// BipertiteGraphConvolution_39573828666026
// MI455X (gfx1250) — matrix operand order repaired and checked by reading, not compiled, not run on hardware
//
#include <hip/hip_runtime.h>

typedef __attribute__((ext_vector_type(16))) _Float16 v16h;
typedef __attribute__((ext_vector_type(8)))  _Float16 v8h;
typedef __attribute__((ext_vector_type(8)))  float    v8f;

#define NU 8192
#define NV 8192
#define FD 64
#define NR 5

// ---------------------------------------------------------------------------
// Phase 1: P[side][r][o][node] = sum_f feat[node][f] * W[r][f][o]   (store f16)
//   side 0: feat = v_feature, W = weight_u   (feeds out_u)
//   side 1: feat = u_feature, W = weight_v   (feeds out_v)
// Node-contiguous layout so phase-2 B tiles are contiguous 64B rows.
// ---------------------------------------------------------------------------
__global__ void proj_kernel(const float* __restrict__ u_feature,
                            const float* __restrict__ v_feature,
                            const float* __restrict__ weight_u,
                            const float* __restrict__ weight_v,
                            _Float16* __restrict__ P)
{
    const int side = blockIdx.y;
    const int idx  = blockIdx.x * blockDim.x + threadIdx.x;   // [0, 5*64*8192)
    const int node = idx & (NV - 1);
    const int o    = (idx >> 13) & 63;
    const int r    = idx >> 19;
    const float* feat = (side == 0) ? v_feature : u_feature;
    const float* W    = (side == 0) ? weight_u  : weight_v;
    float acc = 0.f;
    #pragma unroll 8
    for (int f = 0; f < FD; ++f)
        acc = fmaf(feat[node * FD + f], W[(r * FD + f) * FD + o], acc);
    P[(size_t)((side * NR + r) * FD + o) * NV + node] = (_Float16)acc;
}

// ---------------------------------------------------------------------------
// Async cache->LDS DMA of one 32-K B slab: P[., ., k0:k0+32] (5*64*32 f16).
// 1280 chunks of 16B; each of 256 threads issues 5 global_load_async_to_lds.
// Generic pointer low 32 bits == LDS byte offset (flat LDS aperture rule).
// ---------------------------------------------------------------------------
__device__ __forceinline__ void async_fill(const _Float16* __restrict__ P,
                                           _Float16* ldsbase, int k0, int tid)
{
    #pragma unroll
    for (int c = 0; c < 5; ++c) {
        const int idx = tid + c * 256;            // 0..1279
        const int kq  = idx & 3;                  // 16B quarter of 64B row
        const int o   = (idx >> 2) & 63;
        const int r   = idx >> 8;
        const unsigned long long ga =
            (unsigned long long)(size_t)(P + (size_t)(r * FD + o) * NV + k0 + kq * 8);
        const unsigned lds =
            (unsigned)(size_t)(ldsbase + (r * FD + o) * 32 + kq * 8);
        asm volatile("global_load_async_to_lds_b128 %0, %1, off"
                     :: "v"(lds), "v"(ga) : "memory");
    }
}

// B-operand fetch from LDS: lane L -> N = L%16, halfs h -> K = (h/8)*16 + 8*(L>=16) + (h%8)
__device__ __forceinline__ v16h load_b(const _Float16* Bb, int t, int lm, int hi)
{
    const int r = t >> 2, j = t & 3;
    const _Float16* bp = Bb + (r * FD + j * 16 + lm) * 32 + hi * 8;
    const v8h b0 = *(const v8h*)bp;
    const v8h b1 = *(const v8h*)(bp + 16);
    return __builtin_shufflevector(b0, b1,
        0,1,2,3,4,5,6,7,8,9,10,11,12,13,14,15);
}

// ---------------------------------------------------------------------------
// Phase 2: out[side] = relu( diag(1/deg) * sum_r M_r @ P_r ), f16 WMMA.
//   256 threads = 8 waves; wave w owns rows m0 = bx*128 + w*16; waves share
//   the K loop so B slabs are staged once per 32-K step (async DMA, double-
//   buffered, one barrier per step). A-operand = per-rating one-hot masks,
//   exact in f16, built to the CDNA5 16-bit 16x32 A layout:
//     lane L: M = L%16 ; half h: K = (h/8)*16 + 8*(L>=16) + (h%8)
// ---------------------------------------------------------------------------
template<int SIDE>
__global__ void bgc_main(const int* __restrict__ adj,
                         const _Float16* __restrict__ Pall,
                         float* __restrict__ out_all)
{
    __shared__ __align__(16) _Float16 Blds[2][NR * 64 * 32];  // 2 x 20 KB
    __shared__ int cntlds[256];

    const int tid  = threadIdx.x;
    const int lane = tid & 31;
    const int wave = tid >> 5;
    const int lm   = lane & 15;
    const int hi   = lane >> 4;
    const int m0   = blockIdx.x * 128 + wave * 16;
    const int row  = m0 + lm;

    const _Float16* P = Pall + (size_t)SIDE * NR * FD * NV;

    v8f acc[4] = {};
    int cnt = 0;

    async_fill(P, &Blds[0][0], 0, tid);          // prologue fill, buf 0
    int buf = 0;

    for (int k0 = 0; k0 < NV; k0 += 32) {
        // ---- this lane's 16 adj values (A-layout order) ----
        int vals[16];
        if (SIDE == 0) {
            // two runs of 8 consecutive ints per lane -> 4 x b128
            const int* rowp = adj + (size_t)row * NV + k0 + hi * 8;
            const int4 t0 = *(const int4*)(rowp);
            const int4 t1 = *(const int4*)(rowp + 4);
            const int4 t2 = *(const int4*)(rowp + 16);
            const int4 t3 = *(const int4*)(rowp + 20);
            vals[0]=t0.x;  vals[1]=t0.y;  vals[2]=t0.z;  vals[3]=t0.w;
            vals[4]=t1.x;  vals[5]=t1.y;  vals[6]=t1.z;  vals[7]=t1.w;
            vals[8]=t2.x;  vals[9]=t2.y;  vals[10]=t2.z; vals[11]=t2.w;
            vals[12]=t3.x; vals[13]=t3.y; vals[14]=t3.z; vals[15]=t3.w;
            __builtin_prefetch(rowp + 32, 0, 1);
        } else {
            // transposed walk: K indexes adj rows; half-wave = 64B segments
            #pragma unroll
            for (int h = 0; h < 16; ++h) {
                const int K = (h >> 3) * 16 + hi * 8 + (h & 7);
                vals[h] = adj[(size_t)(k0 + K) * NV + row];
            }
        }

        // ---- one-hot f16 masks (exact) + degree count ----
        v16h a[NR];
        #pragma unroll
        for (int h = 0; h < 16; ++h) {
            const int w = vals[h];
            cnt += (w != 0);
            #pragma unroll
            for (int r = 0; r < NR; ++r)
                a[r][h] = (w == r + 1) ? (_Float16)1.0f : (_Float16)0.0f;
        }

        // ---- wait for our slab DMA, sync, kick off next slab's DMA ----
        asm volatile("s_wait_asynccnt 0" ::: "memory");
        __syncthreads();
        if (k0 + 32 < NV)
            async_fill(P, &Blds[buf ^ 1][0], k0 + 32, tid);

        // ---- 20 WMMAs, B register loads pipelined one tile ahead ----
        const _Float16* Bb = &Blds[buf][0];
        v16h bcur = load_b(Bb, 0, lm, hi);
        #pragma unroll
        for (int t = 0; t < 20; ++t) {
            const v16h bnxt = (t < 19) ? load_b(Bb, t + 1, lm, hi) : bcur;
            acc[t & 3] = __builtin_amdgcn_wmma_f32_16x16x32_f16(
                false, a[t >> 2], false, bcur, (short)0, acc[t & 3],
                false, false);
            bcur = bnxt;
        }
        buf ^= 1;
    }

    // ---- degree reduction (lanes L, L+16 cover one row), scale, relu ----
    __syncthreads();
    cntlds[tid] = cnt;
    __syncthreads();

    float* out = out_all + (size_t)SIDE * NU * FD;
    #pragma unroll
    for (int i = 0; i < 8; ++i) {
        const int m = i + hi * 8;                 // C-layout M for VGPR i
        const int d = cntlds[wave * 32 + m] + cntlds[wave * 32 + m + 16];
        const float s = (d > 0) ? 1.0f / (float)d : 0.0f;
        #pragma unroll
        for (int j = 0; j < 4; ++j) {
            const float x = acc[j][i] * s;
            out[(size_t)(m0 + m) * FD + j * 16 + lm] = (x > 0.f) ? x : 0.f;
        }
    }
}

// ---------------------------------------------------------------------------
extern "C" void kernel_launch(void* const* d_in, const int* in_sizes, int n_in,
                              void* d_out, int out_size, void* d_ws, size_t ws_size,
                              hipStream_t stream) {
    const int*   adj = (const int*)d_in[0];
    const float* uf  = (const float*)d_in[1];
    const float* vf  = (const float*)d_in[2];
    const float* wu  = (const float*)d_in[3];
    const float* wv  = (const float*)d_in[4];

    _Float16* P  = (_Float16*)d_ws;    // 2 sides * 5 * 64 * 8192 f16 = 10.5 MB
    float*   out = (float*)d_out;      // [out_u | out_v], each 8192*64 f32

    dim3 g1(NR * FD * NV / 256, 2);
    proj_kernel<<<g1, 256, 0, stream>>>(uf, vf, wu, wv, P);

    bgc_main<0><<<dim3(NU / 128), 256, 0, stream>>>(adj, P, out);
    bgc_main<1><<<dim3(NV / 128), 256, 0, stream>>>(adj, P, out);
}
